// GPT_TIT_63539746177122
// MI455X (gfx1250) — matrix operand order repaired and checked by reading, not compiled, not run on hardware
//
#include <hip/hip_runtime.h>

// ---------------- CDNA5 WMMA types ----------------
typedef __attribute__((ext_vector_type(16))) __bf16 v16bf;
typedef __attribute__((ext_vector_type(8)))  __bf16 v8bf;
typedef __attribute__((ext_vector_type(8)))  float  v8f;

union Frag16 { v16bf v; v8bf h[2]; uint4 u[2]; };
union AccU   { v8f v; float f[8]; };

__device__ __forceinline__ unsigned short f2bf(float f) {
  union { __bf16 h; unsigned short u; } r;
  r.h = (__bf16)f;                       // native f32->bf16 (RNE)
  return r.u;
}

// ---------------- constants ----------------
#define BSZ   64
#define TT    30
#define BT    1920          // BSZ*TT
#define E     128
#define CH    28224         // 4*84*84
#define LT    89            // 3*T - 1
#define MTOK  5696          // BSZ*LT
#define NHEAD 8
#define HD    16
#define L4    512
#define NLAY  6
#define NVOC  18
#define KSPLIT 21           // split-K for patch GEMM: 28224/21 = 1344 (42 k-blocks, even)
#define KS     1344

// ---------------- fragment load helpers ----------------
// A fragment (bf16 source): ISA 16-bit A layout, lane half g holds K={g*8+0..7, g*8+16..23}
__device__ __forceinline__ void load_a_bf(const unsigned short* __restrict__ arow,
                                          int kb, Frag16& a) {
  a.u[0] = *(const uint4*)(arow + kb);
  a.u[1] = *(const uint4*)(arow + kb + 16);
}
// A fragment from f32 source, converted to bf16 with native cvt (RNE)
__device__ __forceinline__ void load_a_f32(const float* __restrict__ arow,
                                           int kb, Frag16& a) {
  const float4* ap0 = (const float4*)(arow + kb);
  const float4* ap1 = (const float4*)(arow + kb + 16);
  float4 f0 = ap0[0], f1 = ap0[1], f2 = ap1[0], f3 = ap1[1];
  v8f lo = {f0.x, f0.y, f0.z, f0.w, f1.x, f1.y, f1.z, f1.w};
  v8f hi = {f2.x, f2.y, f2.z, f2.w, f3.x, f3.y, f3.z, f3.w};
  a.h[0] = __builtin_convertvector(lo, v8bf);
  a.h[1] = __builtin_convertvector(hi, v8bf);
}
// 4 B fragments (columns nbase..nbase+63), lane half g holds K = kb+{g*8+0..7, g*8+16..23}
__device__ __forceinline__ void load_b4(const unsigned short* __restrict__ W, int K,
                                        int nbase, int li, int g, int kb,
                                        Frag16 b[4]) {
#pragma unroll
  for (int nt = 0; nt < 4; nt++) {
    const uint4* bp = (const uint4*)(W + (size_t)(nbase + nt * 16 + li) * K + kb + g * 8);
    b[nt].u[0] = bp[0];
    b[nt].u[1] = bp[2];
  }
}
__device__ __forceinline__ void wmma4(const Frag16& a, const Frag16 b[4], v8f acc[4]) {
#pragma unroll
  for (int nt = 0; nt < 4; nt++)
    acc[nt] = __builtin_amdgcn_wmma_f32_16x16x32_bf16(
        false, a.v, false, b[nt].v, (short)0, acc[nt], false, false);
}

// =====================================================================
// f32 -> bf16 convert (weights)
// =====================================================================
__global__ void cvt_bf16_kernel(const float* __restrict__ s,
                                unsigned short* __restrict__ d, int n) {
  int i = blockIdx.x * 256 + threadIdx.x;
  if (i < n) d[i] = f2bf(s[i]);
}

// =====================================================================
// Stage 1: patch embedding GEMM (1920 x 28224) @ (28224 x 128)
// grid = (M/16, KSPLIT, N/64); double-buffered software pipeline
// =====================================================================
__global__ __launch_bounds__(32) void patch_gemm_wmma(
    const float* __restrict__ states,
    const unsigned short* __restrict__ wbf,
    float* __restrict__ partial) {
  const int K = CH;
  const v8f vzero = {0.f,0.f,0.f,0.f,0.f,0.f,0.f,0.f};
  int lane = threadIdx.x;
  int li = lane & 15, g = lane >> 4;
  int mbase = blockIdx.x * 16;
  int k0 = blockIdx.y * KS;
  int nbase = blockIdx.z * 64;
  v8f acc[4];
#pragma unroll
  for (int i = 0; i < 4; i++) acc[i] = vzero;

  const float* arow = states + (size_t)(mbase + li) * K + g * 8;

  Frag16 a0, a1, b0[4], b1[4];
  load_a_f32(arow, k0, a0);
  load_b4(wbf, K, nbase, li, g, k0, b0);
  int kb = k0;
  const int HALF = KS / 64;  // 21 double-iterations
  for (int it = 0; it < HALF; ++it) {
    __builtin_prefetch(arow + kb + 64, 0, 3);          // stream next A k-blocks
    load_a_f32(arow, kb + 32, a1);                     // fill buffer 1
    load_b4(wbf, K, nbase, li, g, kb + 32, b1);
    wmma4(a0, b0, acc);                                // consume buffer 0
    if (it + 1 < HALF) {
      load_a_f32(arow, kb + 64, a0);                   // refill buffer 0
      load_b4(wbf, K, nbase, li, g, kb + 64, b0);
    }
    wmma4(a1, b1, acc);                                // consume buffer 1
    kb += 64;
  }

  float* outp = partial + (size_t)blockIdx.y * BT * E;
#pragma unroll
  for (int nt = 0; nt < 4; nt++) {
    AccU t; t.v = acc[nt];
#pragma unroll
    for (int r = 0; r < 8; r++) {
      int m = mbase + r + 8 * g;
      int n = nbase + nt * 16 + li;
      outp[(size_t)m * E + n] = t.f[r];
    }
  }
}

__global__ void reduce_patch_kernel(const float* __restrict__ part,
                                    float* __restrict__ patch, int n) {
  int i = blockIdx.x * 256 + threadIdx.x;
  if (i < n) {
    float s = 0.f;
#pragma unroll
    for (int j = 0; j < KSPLIT; j++) s += part[(size_t)j * n + i];
    patch[i] = s;
  }
}

// =====================================================================
// Generic bf16 WMMA GEMM: out(MxN) = A(MxK) @ W(NxK)^T + bias [+resid] [gelu]
// grid = (M/16, N/64); double-buffered software pipeline
// =====================================================================
__global__ __launch_bounds__(32) void gemm_bf16_wmma(
    const unsigned short* __restrict__ A,
    const unsigned short* __restrict__ W,
    const float* __restrict__ bias,
    const float* __restrict__ resid,
    float* __restrict__ outF,
    unsigned short* __restrict__ outB,
    int M, int N, int K, int act) {
  const v8f vzero = {0.f,0.f,0.f,0.f,0.f,0.f,0.f,0.f};
  int lane = threadIdx.x;
  int li = lane & 15, g = lane >> 4;
  int mbase = blockIdx.x * 16;
  int nbase = blockIdx.y * 64;
  v8f acc[4];
#pragma unroll
  for (int i = 0; i < 4; i++) acc[i] = vzero;

  const unsigned short* arow = A + (size_t)(mbase + li) * K + g * 8;

  Frag16 a0, a1, b0[4], b1[4];
  load_a_bf(arow, 0, a0);
  load_b4(W, K, nbase, li, g, 0, b0);
  int kb = 0;
  const int HALF = K / 64;   // K is a multiple of 64 (128 or 512)
  for (int it = 0; it < HALF; ++it) {
    __builtin_prefetch(arow + kb + 64, 0, 3);
    load_a_bf(arow, kb + 32, a1);
    load_b4(W, K, nbase, li, g, kb + 32, b1);
    wmma4(a0, b0, acc);
    if (it + 1 < HALF) {
      load_a_bf(arow, kb + 64, a0);
      load_b4(W, K, nbase, li, g, kb + 64, b0);
    }
    wmma4(a1, b1, acc);
    kb += 64;
  }

#pragma unroll
  for (int nt = 0; nt < 4; nt++) {
    AccU t; t.v = acc[nt];
#pragma unroll
    for (int r = 0; r < 8; r++) {
      int m = mbase + r + 8 * g;
      int n = nbase + nt * 16 + li;
      float v = t.f[r] + bias[n];
      if (resid) v += resid[(size_t)m * N + n];
      if (act == 1) v = 0.5f * v * (1.0f + erff(v * 0.70710678118f));  // exact GELU
      if (outF) outF[(size_t)m * N + n] = v;
      if (outB) outB[(size_t)m * N + n] = f2bf(v);
    }
  }
}

// =====================================================================
// LayerNorm over E=128, writes bf16 (feeds WMMA GEMMs)
// =====================================================================
__global__ __launch_bounds__(128) void ln_bf16_kernel(
    const float* __restrict__ x, const float* __restrict__ gm,
    const float* __restrict__ bt, unsigned short* __restrict__ o) {
  __shared__ float red[128];
  int t = threadIdx.x;
  size_t row = blockIdx.x;
  float v = x[row * E + t];
  red[t] = v; __syncthreads();
  for (int s = 64; s; s >>= 1) { if (t < s) red[t] += red[t + s]; __syncthreads(); }
  float mean = red[0] * (1.f / 128.f); __syncthreads();
  float d = v - mean;
  red[t] = d * d; __syncthreads();
  for (int s = 64; s; s >>= 1) { if (t < s) red[t] += red[t + s]; __syncthreads(); }
  float var = red[0] * (1.f / 128.f);
  o[row * E + t] = f2bf(d * rsqrtf(var + 1e-5f) * gm[t] + bt[t]);
}

// =====================================================================
// Inner TIT block (2 tokens x 128): LN -> 1-head MHA -> LN -> ReLU FFN
// =====================================================================
__device__ inline float blk_sum(float v, float* red) {
  int t = threadIdx.x;
  __syncthreads();
  red[t] = v; __syncthreads();
  for (int s = 64; s; s >>= 1) { if (t < s) red[t] += red[t + s]; __syncthreads(); }
  float r = red[0]; __syncthreads();
  return r;
}

__global__ __launch_bounds__(128) void inner_tit_kernel(
    const float* __restrict__ patch, const float* __restrict__ cls,
    const float* __restrict__ ipw, const float* __restrict__ ipb,
    const float* __restrict__ ow,  const float* __restrict__ ob,
    const float* __restrict__ l1g, const float* __restrict__ l1b,
    const float* __restrict__ l2g, const float* __restrict__ l2b,
    const float* __restrict__ fw1, const float* __restrict__ fb1,
    const float* __restrict__ fw2, const float* __restrict__ fb2,
    float* __restrict__ state_emb) {
  __shared__ float xs[2][128], hs[2][128], qs[2][128], ks[2][128], vs[2][128];
  __shared__ float red[128], sc[4];
  int t = threadIdx.x, bt = blockIdx.x;
  xs[0][t] = cls[t];
  xs[1][t] = patch[(size_t)bt * E + t];
  __syncthreads();
  for (int r = 0; r < 2; r++) {                                  // LN1
    float v = xs[r][t];
    float mean = blk_sum(v, red) * (1.f / 128.f);
    float d = v - mean;
    float var = blk_sum(d * d, red) * (1.f / 128.f);
    hs[r][t] = d * rsqrtf(var + 1e-5f) * l1g[t] + l1b[t];
  }
  __syncthreads();
  for (int r = 0; r < 2; r++) {                                  // QKV
    float q = ipb[t], k = ipb[128 + t], v = ipb[256 + t];
    for (int e = 0; e < 128; e++) {
      float h = hs[r][e];
      q += h * ipw[(size_t)t * E + e];
      k += h * ipw[(size_t)(128 + t) * E + e];
      v += h * ipw[(size_t)(256 + t) * E + e];
    }
    qs[r][t] = q; ks[r][t] = k; vs[r][t] = v;
  }
  __syncthreads();
  if (t < 4) {
    int r = t >> 1, c = t & 1;
    float s = 0.f;
    for (int e = 0; e < 128; e++) s += qs[r][e] * ks[c][e];
    sc[t] = s * 0.08838834764831845f;   // 1/sqrt(128)
  }
  __syncthreads();
  if (t < 2) {
    float a = sc[t * 2], b = sc[t * 2 + 1];
    float m = fmaxf(a, b);
    float ea = expf(a - m), eb = expf(b - m);
    float inv = 1.f / (ea + eb);
    sc[t * 2] = ea * inv; sc[t * 2 + 1] = eb * inv;
  }
  __syncthreads();
  for (int r = 0; r < 2; r++)
    hs[r][t] = sc[r * 2] * vs[0][t] + sc[r * 2 + 1] * vs[1][t];  // att @ v
  __syncthreads();
  for (int r = 0; r < 2; r++) {                                  // out proj
    float y = ob[t];
    for (int e = 0; e < 128; e++) y += hs[r][e] * ow[(size_t)t * E + e];
    qs[r][t] = y;
  }
  __syncthreads();
  for (int r = 0; r < 2; r++) xs[r][t] += qs[r][t];              // residual
  __syncthreads();
  for (int r = 0; r < 2; r++) {                                  // LN2
    float v = xs[r][t];
    float mean = blk_sum(v, red) * (1.f / 128.f);
    float d = v - mean;
    float var = blk_sum(d * d, red) * (1.f / 128.f);
    hs[r][t] = d * rsqrtf(var + 1e-5f) * l2g[t] + l2b[t];
  }
  __syncthreads();
  for (int r = 0; r < 2; r++) {                                  // FFN1 + ReLU
    float f = fb1[t];
    for (int e = 0; e < 128; e++) f += hs[r][e] * fw1[(size_t)t * E + e];
    qs[r][t] = fmaxf(f, 0.f);
  }
  __syncthreads();
  for (int r = 0; r < 2; r++) {                                  // FFN2
    float o = fb2[t];
    for (int e = 0; e < 128; e++) o += qs[r][e] * fw2[(size_t)t * E + e];
    vs[r][t] = o;
  }
  __syncthreads();
  state_emb[(size_t)bt * E + t] = xs[0][t] + vs[0][t];           // class token
}

// =====================================================================
// Token interleave + positional embeddings -> x (B, 89, 128)
// =====================================================================
__global__ __launch_bounds__(128) void build_tokens_kernel(
    const float* __restrict__ state_emb, const float* __restrict__ rtgs,
    const int* __restrict__ actions, const int* __restrict__ tsteps,
    const float* __restrict__ retw, const float* __restrict__ retb,
    const float* __restrict__ atab, const float* __restrict__ pos,
    const float* __restrict__ gpos, float* __restrict__ x) {
  int e = threadIdx.x;
  int bp = blockIdx.x;
  int b = bp / LT, p = bp % LT;
  int typ = p % 3, tt = p / 3;
  float val;
  if (typ == 0)      val = tanhf(rtgs[b * TT + tt] * retw[e] + retb[e]);
  else if (typ == 1) val = state_emb[(size_t)(b * TT + tt) * E + e];
  else               val = tanhf(atab[(size_t)actions[b * TT + tt + 1] * E + e]);
  int ts = tsteps[b];
  val += gpos[(size_t)ts * E + e] + pos[(size_t)p * E + e];
  x[(size_t)bp * E + e] = val;
}

// =====================================================================
// Outer causal attention per (b,h): 89x16 head, 2-pass softmax via LDS
// =====================================================================
__global__ __launch_bounds__(96) void attn_outer_kernel(
    const float* __restrict__ q, const float* __restrict__ k,
    const float* __restrict__ v, unsigned short* __restrict__ ybf) {
  __shared__ float ksh[LT][HD], vsh[LT][HD];
  int t = threadIdx.x;
  int bh = blockIdx.x;
  int b = bh >> 3, h = bh & 7;
  size_t base = (size_t)b * LT * E + h * HD;
  for (int idx = t; idx < LT * HD; idx += 96) {
    int j = idx >> 4, d = idx & 15;
    ksh[j][d] = k[base + (size_t)j * E + d];
    vsh[j][d] = v[base + (size_t)j * E + d];
  }
  __syncthreads();
  if (t < LT) {
    float qr[HD];
#pragma unroll
    for (int d = 0; d < HD; d++) qr[d] = q[base + (size_t)t * E + d];
    float m = -1e30f;
    for (int j = 0; j <= t; j++) {
      float s = 0.f;
#pragma unroll
      for (int d = 0; d < HD; d++) s += qr[d] * ksh[j][d];
      m = fmaxf(m, s * 0.25f);                     // 1/sqrt(16)
    }
    float l = 0.f, acc[HD];
#pragma unroll
    for (int d = 0; d < HD; d++) acc[d] = 0.f;
    for (int j = 0; j <= t; j++) {
      float s = 0.f;
#pragma unroll
      for (int d = 0; d < HD; d++) s += qr[d] * ksh[j][d];
      float w = expf(s * 0.25f - m);
      l += w;
#pragma unroll
      for (int d = 0; d < HD; d++) acc[d] += w * vsh[j][d];
    }
    float inv = 1.f / l;
#pragma unroll
    for (int d = 0; d < HD; d++)
      ybf[base + (size_t)t * E + d] = f2bf(acc[d] * inv);
  }
}

// =====================================================================
// Final LN + head at state positions (p = 1 + 3*tp) -> (B, 30, 18) f32
// =====================================================================
__global__ __launch_bounds__(128) void head_kernel(
    const float* __restrict__ x, const float* __restrict__ g,
    const float* __restrict__ bta, const float* __restrict__ hw,
    float* __restrict__ out) {
  __shared__ float red[128], lnv[128];
  int t = threadIdx.x;
  int bi = blockIdx.x;
  int b = bi / TT, tp = bi % TT;
  size_t row = (size_t)b * LT + 1 + 3 * tp;
  float v = x[row * E + t];
  red[t] = v; __syncthreads();
  for (int s = 64; s; s >>= 1) { if (t < s) red[t] += red[t + s]; __syncthreads(); }
  float mean = red[0] * (1.f / 128.f); __syncthreads();
  float d = v - mean;
  red[t] = d * d; __syncthreads();
  for (int s = 64; s; s >>= 1) { if (t < s) red[t] += red[t + s]; __syncthreads(); }
  float var = red[0] * (1.f / 128.f); __syncthreads();
  lnv[t] = d * rsqrtf(var + 1e-5f) * g[t] + bta[t];
  __syncthreads();
  if (t < NVOC) {
    float s = 0.f;
    for (int e = 0; e < 128; e++) s += lnv[e] * hw[(size_t)t * E + e];
    out[(size_t)bi * NVOC + t] = s;
  }
}

// =====================================================================
extern "C" void kernel_launch(void* const* d_in, const int* in_sizes, int n_in,
                              void* d_out, int out_size, void* d_ws, size_t ws_size,
                              hipStream_t stream) {
  (void)in_sizes; (void)n_in; (void)out_size; (void)ws_size;
  const float* states = (const float*)d_in[0];
  const int*   actions= (const int*)  d_in[1];
  const float* rtgs   = (const float*)d_in[2];
  const int*   tsteps = (const int*)  d_in[3];
  const float* conv_w = (const float*)d_in[4];
  const float* cls    = (const float*)d_in[5];
  const float* ipw =(const float*)d_in[6];  const float* ipb =(const float*)d_in[7];
  const float* ow  =(const float*)d_in[8];  const float* ob  =(const float*)d_in[9];
  const float* il1g=(const float*)d_in[10]; const float* il1b=(const float*)d_in[11];
  const float* il2g=(const float*)d_in[12]; const float* il2b=(const float*)d_in[13];
  const float* fw1 =(const float*)d_in[14]; const float* fb1 =(const float*)d_in[15];
  const float* fw2 =(const float*)d_in[16]; const float* fb2 =(const float*)d_in[17];
  const float* retw=(const float*)d_in[18]; const float* retb=(const float*)d_in[19];
  const float* atab=(const float*)d_in[20];
  const float* pos =(const float*)d_in[21]; const float* gpos=(const float*)d_in[22];
  const float* ln1g=(const float*)d_in[23]; const float* ln1b=(const float*)d_in[24];
  const float* Wq  =(const float*)d_in[25]; const float* bq  =(const float*)d_in[26];
  const float* Wk  =(const float*)d_in[27]; const float* bk  =(const float*)d_in[28];
  const float* Wv  =(const float*)d_in[29]; const float* bv  =(const float*)d_in[30];
  const float* Wo  =(const float*)d_in[31]; const float* bo  =(const float*)d_in[32];
  const float* ln2g=(const float*)d_in[33]; const float* ln2b=(const float*)d_in[34];
  const float* W1  =(const float*)d_in[35]; const float* b1  =(const float*)d_in[36];
  const float* W2  =(const float*)d_in[37]; const float* b2  =(const float*)d_in[38];
  const float* lnfg=(const float*)d_in[39]; const float* lnfb=(const float*)d_in[40];
  const float* hw  =(const float*)d_in[41];
  float* out = (float*)d_out;

  // ---- workspace bump allocator ----
  char* ws = (char*)d_ws;
  size_t off = 0;
  auto take = [&](size_t bytes) -> char* {
    char* p = ws + off;
    off = (off + bytes + 255) & ~(size_t)255;
    return p;
  };
  unsigned short* conv_bf = (unsigned short*)take((size_t)E * CH * 2);        // 7.2 MB
  unsigned short* wq_bf   = (unsigned short*)take((size_t)NLAY * E * E * 2);
  unsigned short* wk_bf   = (unsigned short*)take((size_t)NLAY * E * E * 2);
  unsigned short* wv_bf   = (unsigned short*)take((size_t)NLAY * E * E * 2);
  unsigned short* wo_bf   = (unsigned short*)take((size_t)NLAY * E * E * 2);
  unsigned short* w1_bf   = (unsigned short*)take((size_t)NLAY * L4 * E * 2);
  unsigned short* w2_bf   = (unsigned short*)take((size_t)NLAY * E * L4 * 2);
  float*          kpart   = (float*)take((size_t)KSPLIT * BT * E * 4);        // 20.6 MB
  float*          patch   = (float*)take((size_t)BT * E * 4);
  float*          semb    = (float*)take((size_t)BT * E * 4);
  float*          x       = (float*)take((size_t)MTOK * E * 4);
  unsigned short* hh_bf   = (unsigned short*)take((size_t)MTOK * E * 2);
  float*          qf      = (float*)take((size_t)MTOK * E * 4);
  float*          kf      = (float*)take((size_t)MTOK * E * 4);
  float*          vf      = (float*)take((size_t)MTOK * E * 4);
  unsigned short* y_bf    = (unsigned short*)take((size_t)MTOK * E * 2);
  unsigned short* h1_bf   = (unsigned short*)take((size_t)MTOK * L4 * 2);

  auto cvt = [&](const float* s, unsigned short* d, int n) {
    cvt_bf16_kernel<<<(n + 255) / 256, 256, 0, stream>>>(s, d, n);
  };
  // ---- weight conversions ----
  cvt(conv_w, conv_bf, E * CH);
  cvt(Wq, wq_bf, NLAY * E * E);
  cvt(Wk, wk_bf, NLAY * E * E);
  cvt(Wv, wv_bf, NLAY * E * E);
  cvt(Wo, wo_bf, NLAY * E * E);
  cvt(W1, w1_bf, NLAY * L4 * E);
  cvt(W2, w2_bf, NLAY * E * L4);

  // ---- stage 1: patch embed GEMM, split-K x21, 5040 waves + reduce ----
  patch_gemm_wmma<<<dim3(BT / 16, KSPLIT, 2), 32, 0, stream>>>(states, conv_bf, kpart);
  reduce_patch_kernel<<<(BT * E + 255) / 256, 256, 0, stream>>>(kpart, patch, BT * E);

  // ---- stage 2: inner TIT per (b,t) ----
  inner_tit_kernel<<<BT, 128, 0, stream>>>(patch, cls, ipw, ipb, ow, ob,
                                           il1g, il1b, il2g, il2b,
                                           fw1, fb1, fw2, fb2, semb);

  // ---- stage 3: token interleave + positions ----
  build_tokens_kernel<<<BSZ * LT, 128, 0, stream>>>(semb, rtgs, actions, tsteps,
                                                    retw, retb, atab, pos, gpos, x);

  // ---- stage 4: outer GPT blocks ----
  const int MT = MTOK / 16;  // 356
  for (int i = 0; i < NLAY; i++) {
    ln_bf16_kernel<<<MTOK, 128, 0, stream>>>(x, ln1g + i * E, ln1b + i * E, hh_bf);
    gemm_bf16_wmma<<<dim3(MT, 2), 32, 0, stream>>>(hh_bf, wq_bf + (size_t)i * E * E,
        bq + i * E, nullptr, qf, nullptr, MTOK, E, E, 0);
    gemm_bf16_wmma<<<dim3(MT, 2), 32, 0, stream>>>(hh_bf, wk_bf + (size_t)i * E * E,
        bk + i * E, nullptr, kf, nullptr, MTOK, E, E, 0);
    gemm_bf16_wmma<<<dim3(MT, 2), 32, 0, stream>>>(hh_bf, wv_bf + (size_t)i * E * E,
        bv + i * E, nullptr, vf, nullptr, MTOK, E, E, 0);
    attn_outer_kernel<<<BSZ * NHEAD, 96, 0, stream>>>(qf, kf, vf, y_bf);
    gemm_bf16_wmma<<<dim3(MT, 2), 32, 0, stream>>>(y_bf, wo_bf + (size_t)i * E * E,
        bo + i * E, x, x, nullptr, MTOK, E, E, 0);              // fused residual
    ln_bf16_kernel<<<MTOK, 128, 0, stream>>>(x, ln2g + i * E, ln2b + i * E, hh_bf);
    gemm_bf16_wmma<<<dim3(MT, 8), 32, 0, stream>>>(hh_bf, w1_bf + (size_t)i * L4 * E,
        b1 + i * L4, nullptr, nullptr, h1_bf, MTOK, L4, E, 1);  // fused GELU -> bf16
    gemm_bf16_wmma<<<dim3(MT, 2), 32, 0, stream>>>(h1_bf, w2_bf + (size_t)i * E * L4,
        b2 + i * E, x, x, nullptr, MTOK, E, L4, 0);             // fused residual
  }

  // ---- stage 5: final LN + head at state positions ----
  head_kernel<<<BSZ * TT, 128, 0, stream>>>(x, lnfg, lnfb, hw, out);
}
